// SwinAttention1D_32899449487948
// MI455X (gfx1250) — matrix operand order repaired and checked by reading, not compiled, not run on hardware
//
#include <hip/hip_runtime.h>
#include <hip/hip_bf16.h>

// ---------------------------------------------------------------------------
// Fused shifted-window (Swin-1D) attention block for MI455X / gfx1250.
//   prep kernel : repack w_qkv/w_out fp32 -> bf16 in WMMA B-fragment order (d_ws)
//   main kernel : roll(-4) -> windows(8) -> QKV GEMM (bf16 WMMA) -> attention
//                 -> out-proj GEMM (bf16 WMMA) -> +bias +residual -> LayerNorm
//                 -> roll(+4) -> store.   All intermediates in LDS.
// HBM traffic = x in + y out (536 MB) ~= 23 us floor at 23.3 TB/s; weights are
// L2-resident; bf16 WMMA keeps compute well under the memory floor.
// ---------------------------------------------------------------------------

typedef __attribute__((ext_vector_type(16))) __bf16 v16bf;
typedef __attribute__((ext_vector_type(8)))  float  v8f;

#define NHEAD    4
#define SHIFT_   4
#define CH       256
#define HD       64
#define NSEQ     8192
#define MTOK     64        // tokens per workgroup (= 8 windows)

// LDS strides padded for bank-conflict-free column access (64 banks x 4B)
#define XF_STR   260       // fp32 residual tile   (floats)
#define XB_STR   264       // bf16 activation tile (elems)
#define QK_STR   776       // bf16 qkv tile        (elems)
#define YF_STR   260       // fp32 pre-LN tile     (floats)

#define OFF_XF   0                     // 64*260*4 = 66560
#define OFF_XB   66560                 // 64*264*2 = 33792
#define OFF_QKV  100352                // 64*776*2 = 99328 (aliased as Y f32)
#define OFF_SC   199680                // 2048*4   = 8192
#define SMEM_BYTES 207872

// packed-weight workspace layout (uint = bf16 pair)
#define QKV_PAIRS  98304               // 48 ntiles * 8 ksteps * 32 lanes * 8
#define WOUT_PAIRS 32768               // 16 ntiles * 8 ksteps * 32 lanes * 8

__device__ __forceinline__ unsigned short f2bf(float f) {
  union { float f; unsigned u; } v; v.f = f;
  unsigned r = v.u + 0x7FFFu + ((v.u >> 16) & 1u);   // round-to-nearest-even
  return (unsigned short)(r >> 16);
}
__device__ __forceinline__ float bf2f(unsigned u16) {          // low 16 bits
  union { unsigned u; float f; } v; v.u = u16 << 16; return v.f;
}

// A-fragment (16x32 bf16) from an LDS bf16 tile, assembled by raw bit-cast.
// lanes 0-15 : row M = lane,    K = 0..15  (V0-7)
// lanes16-31 : row M = lane-16, K = 16..31 (V0-7)
__device__ __forceinline__ v16bf load_afrag(const unsigned short* base, int stride,
                                            int m, int kb, int half) {
  union { unsigned u[8]; v16bf v; } r;
  const unsigned short* row = base + m * stride;
#pragma unroll
  for (int v = 0; v < 8; ++v) {
    const int k0 = kb + half * 16 + 2 * v;
    r.u[v] = *(const unsigned*)(row + k0);
  }
  return r.v;
}

// B-fragment: pre-packed bf16 pairs in fragment order -> two b128 loads.
__device__ __forceinline__ v16bf load_bfrag_pre(const unsigned* wp, int tile, int lane) {
  union { uint4 q[2]; v16bf v; } r;
  const uint4* p = (const uint4*)(wp + ((size_t)tile * 32 + lane) * 8);
  r.q[0] = p[0];
  r.q[1] = p[1];
  return r.v;
}

// ---------------------------------------------------------------------------
// Prep: repack w_qkv (768x256) and w_out (256x256) fp32 -> bf16 pairs in
// B-fragment order: pair index = ((ntile*8 + kstep)*32 + lane)*8 + v
// holds W[ntile*16 + lane%16][kstep*32 + (lane/16)*16 + 2v .. +1].
// ---------------------------------------------------------------------------
__global__ __launch_bounds__(256)
void swin1d_pack_weights(const float* __restrict__ w_qkv,
                         const float* __restrict__ w_out,
                         unsigned* __restrict__ ws) {
  const int g = blockIdx.x * 256 + threadIdx.x;     // 0 .. 131071
  const bool isq = (g < QKV_PAIRS);
  const int gl   = isq ? g : g - QKV_PAIRS;
  const int tile = gl >> 8;                         // ntile*8 + kstep
  const int lane = (gl >> 3) & 31;
  const int v    = gl & 7;
  const int nt   = tile >> 3, k = tile & 7;
  const int n    = nt * 16 + (lane & 15);
  const int kk   = k * 32 + (lane >> 4) * 16 + 2 * v;
  const float* w = isq ? w_qkv : w_out;
  const float2 p = *(const float2*)(w + (size_t)n * CH + kk);
  ws[g] = (unsigned)f2bf(p.x) | ((unsigned)f2bf(p.y) << 16);
}

__global__ __launch_bounds__(256)
void swin1d_fused_kernel(const float* __restrict__ x,
                         const unsigned* __restrict__ wpre,   // packed weights
                         const float* __restrict__ b_qkv,
                         const float* __restrict__ b_out,
                         const float* __restrict__ ln_w,
                         const float* __restrict__ ln_b,
                         float* __restrict__ out) {
  extern __shared__ char smem[];
  float*          Xf  = (float*)(smem + OFF_XF);            // residual (fp32)
  unsigned short* Xb  = (unsigned short*)(smem + OFF_XB);   // x bf16 / attn-out bf16
  unsigned short* QKV = (unsigned short*)(smem + OFF_QKV);  // qkv bf16
  float*          Yf  = (float*)(smem + OFF_QKV);           // alias: pre-LN fp32
  float*          SC  = (float*)(smem + OFF_SC);            // scores [8w][4h][8][8]

  const int bidx = blockIdx.x;
  const int bb   = bidx / (NSEQ / MTOK);
  const int p0   = (bidx % (NSEQ / MTOK)) * MTOK;
  const int tid  = threadIdx.x;
  const int wv   = tid >> 5;
  const int lane = tid & 31;
  const int half = lane >> 4;
  const int lmod = lane & 15;

  // ---- Phase 0: load shifted tile, keep fp32 residual + bf16 copy ----------
  {
    const int r  = tid >> 2;
    const int c0 = (tid & 3) * 64;
    const int n  = (p0 + r + SHIFT_) & (NSEQ - 1);           // roll(-SHIFT)
    const float* src = x + ((size_t)bb * NSEQ + n) * CH + c0;
    float*          df = Xf + r * XF_STR + c0;
    unsigned short* db = Xb + r * XB_STR + c0;
#pragma unroll
    for (int t = 0; t < 64; t += 4) {
      float4 v4 = *(const float4*)(src + t);
      *(float4*)(df + t) = v4;
      *(unsigned*)(db + t)     = (unsigned)f2bf(v4.x) | ((unsigned)f2bf(v4.y) << 16);
      *(unsigned*)(db + t + 2) = (unsigned)f2bf(v4.z) | ((unsigned)f2bf(v4.w) << 16);
    }
  }
  __syncthreads();

  // ---- Phase 1: QKV GEMM  [64x256] @ [256x768]^T  (bf16 WMMA, f32 acc) -----
  for (int ntb = 0; ntb < 6; ++ntb) {
    const int nt = wv + ntb * 8;          // 0..47
    const int nG = nt * 16 + lmod;        // output channel 0..767
    v8f acc0 = (v8f){0.f,0.f,0.f,0.f,0.f,0.f,0.f,0.f};
    v8f acc1 = acc0, acc2 = acc0, acc3 = acc0;
    for (int k = 0; k < 8; ++k) {
      const int kb = k * 32;
      __builtin_prefetch(wpre + ((size_t)(nt * 8 + k + 1) * 32 + lane) * 8, 0, 1);
      v16bf bf = load_bfrag_pre(wpre, nt * 8 + k, lane);
      v16bf a0 = load_afrag(Xb, XB_STR,  0 + lmod, kb, half);
      v16bf a1 = load_afrag(Xb, XB_STR, 16 + lmod, kb, half);
      v16bf a2 = load_afrag(Xb, XB_STR, 32 + lmod, kb, half);
      v16bf a3 = load_afrag(Xb, XB_STR, 48 + lmod, kb, half);
      acc0 = __builtin_amdgcn_wmma_f32_16x16x32_bf16(false, a0, false, bf, (short)0, acc0, false, false);
      acc1 = __builtin_amdgcn_wmma_f32_16x16x32_bf16(false, a1, false, bf, (short)0, acc1, false, false);
      acc2 = __builtin_amdgcn_wmma_f32_16x16x32_bf16(false, a2, false, bf, (short)0, acc2, false, false);
      acc3 = __builtin_amdgcn_wmma_f32_16x16x32_bf16(false, a3, false, bf, (short)0, acc3, false, false);
    }
    const float bias = b_qkv[nG];
    v8f accs[4] = {acc0, acc1, acc2, acc3};
#pragma unroll
    for (int mt = 0; mt < 4; ++mt)
#pragma unroll
      for (int v = 0; v < 8; ++v) {
        const int m = mt * 16 + v + 8 * half;                // D layout row
        QKV[m * QK_STR + nG] = f2bf(accs[mt][v] + bias);
      }
  }
  __syncthreads();

  // ---- Phase 2a: scores = q.k^T / 8  (2048 entries, 8 per thread) ----------
#pragma unroll
  for (int t = 0; t < 8; ++t) {
    const int g    = tid + 256 * t;
    const int win  = g >> 8;
    const int rem  = g & 255;
    const int head = rem >> 6;
    const int qi   = (rem >> 3) & 7;
    const int kj   = rem & 7;
    const unsigned short* qrow = QKV + (win * 8 + qi) * QK_STR + head * HD;
    const unsigned short* krow = QKV + (win * 8 + kj) * QK_STR + CH + head * HD;
    float s = 0.f;
#pragma unroll 8
    for (int d = 0; d < HD; d += 2) {
      const unsigned qp = *(const unsigned*)(qrow + d);
      const unsigned kp = *(const unsigned*)(krow + d);
      s += bf2f(qp & 0xFFFFu) * bf2f(kp & 0xFFFFu);
      s += bf2f(qp >> 16)     * bf2f(kp >> 16);
    }
    SC[g] = s * 0.125f;                                       // 1/sqrt(64)
  }
  __syncthreads();

  // ---- Phase 2b: softmax over each row of 8 (one row per thread) -----------
  {
    float v[8];
#pragma unroll
    for (int j = 0; j < 8; ++j) v[j] = SC[tid * 8 + j];
    float mx = v[0];
#pragma unroll
    for (int j = 1; j < 8; ++j) mx = fmaxf(mx, v[j]);
    float s = 0.f;
#pragma unroll
    for (int j = 0; j < 8; ++j) { v[j] = __expf(v[j] - mx); s += v[j]; }
    const float inv = 1.f / s;
#pragma unroll
    for (int j = 0; j < 8; ++j) SC[tid * 8 + j] = v[j] * inv;
  }
  __syncthreads();

  // ---- Phase 2c: O = attn @ V  -> bf16 into Xb (reuse) ---------------------
  {
    const int tok = tid >> 2;            // 0..63
    const int seg = tid & 3;             // head
    const int win = tok >> 3, qi = tok & 7;
    float aw[8];
    const int sbase = ((win * 4 + seg) * 8 + qi) * 8;
#pragma unroll
    for (int j = 0; j < 8; ++j) aw[j] = SC[sbase + j];
    unsigned short* orow = Xb + tok * XB_STR + seg * HD;
    const unsigned short* vbase = QKV + (win * 8) * QK_STR + 2 * CH + seg * HD;
#pragma unroll 4
    for (int d = 0; d < HD; d += 2) {
      float o0 = 0.f, o1 = 0.f;
#pragma unroll
      for (int j = 0; j < 8; ++j) {
        const unsigned vp = *(const unsigned*)(vbase + j * QK_STR + d);
        o0 += aw[j] * bf2f(vp & 0xFFFFu);
        o1 += aw[j] * bf2f(vp >> 16);
      }
      *(unsigned*)(orow + d) = (unsigned)f2bf(o0) | ((unsigned)f2bf(o1) << 16);
    }
  }
  __syncthreads();

  // ---- Phase 3: out-proj GEMM + bias + residual -> Yf (aliases QKV) --------
  {
    const unsigned* wop = wpre + QKV_PAIRS;
    for (int ntb = 0; ntb < 2; ++ntb) {
      const int nt = wv + ntb * 8;        // 0..15
      const int nG = nt * 16 + lmod;      // channel 0..255
      v8f acc0 = (v8f){0.f,0.f,0.f,0.f,0.f,0.f,0.f,0.f};
      v8f acc1 = acc0, acc2 = acc0, acc3 = acc0;
      for (int k = 0; k < 8; ++k) {
        const int kb = k * 32;
        v16bf bf = load_bfrag_pre(wop, nt * 8 + k, lane);
        v16bf a0 = load_afrag(Xb, XB_STR,  0 + lmod, kb, half);
        v16bf a1 = load_afrag(Xb, XB_STR, 16 + lmod, kb, half);
        v16bf a2 = load_afrag(Xb, XB_STR, 32 + lmod, kb, half);
        v16bf a3 = load_afrag(Xb, XB_STR, 48 + lmod, kb, half);
        acc0 = __builtin_amdgcn_wmma_f32_16x16x32_bf16(false, a0, false, bf, (short)0, acc0, false, false);
        acc1 = __builtin_amdgcn_wmma_f32_16x16x32_bf16(false, a1, false, bf, (short)0, acc1, false, false);
        acc2 = __builtin_amdgcn_wmma_f32_16x16x32_bf16(false, a2, false, bf, (short)0, acc2, false, false);
        acc3 = __builtin_amdgcn_wmma_f32_16x16x32_bf16(false, a3, false, bf, (short)0, acc3, false, false);
      }
      const float bias = b_out[nG];
      v8f accs[4] = {acc0, acc1, acc2, acc3};
#pragma unroll
      for (int mt = 0; mt < 4; ++mt)
#pragma unroll
        for (int v = 0; v < 8; ++v) {
          const int m = mt * 16 + v + 8 * half;
          Yf[m * YF_STR + nG] = accs[mt][v] + bias + Xf[m * XF_STR + nG];
        }
    }
  }
  __syncthreads();

  // ---- Phase 4: LayerNorm over C, inverse roll, store ----------------------
  {
    const int tok = tid & 63;
    const int seg = tid >> 6;            // quarter of the channel dim
    const float* yrow = Yf + tok * YF_STR;
    float s = 0.f, s2 = 0.f;
#pragma unroll 4
    for (int c = 0; c < CH; c += 4) {
      const float4 y = *(const float4*)(yrow + c);
      s  += y.x + y.y + y.z + y.w;
      s2 += y.x * y.x + y.y * y.y + y.z * y.z + y.w * y.w;
    }
    const float mu   = s  * (1.f / CH);
    const float var  = s2 * (1.f / CH) - mu * mu;
    const float rstd = rsqrtf(var + 1e-5f);
    const int n = (p0 + tok + SHIFT_) & (NSEQ - 1);          // roll(+SHIFT)
    float* dst = out + ((size_t)bb * NSEQ + n) * CH + seg * 64;
    const float* yseg = yrow + seg * 64;
#pragma unroll
    for (int t = 0; t < 64; t += 4) {
      float4 y = *(const float4*)(yseg + t);
      float4 w = *(const float4*)(ln_w + seg * 64 + t);
      float4 b = *(const float4*)(ln_b + seg * 64 + t);
      float4 o;
      o.x = (y.x - mu) * rstd * w.x + b.x;
      o.y = (y.y - mu) * rstd * w.y + b.y;
      o.z = (y.z - mu) * rstd * w.z + b.z;
      o.w = (y.w - mu) * rstd * w.w + b.w;
      *(float4*)(dst + t) = o;
    }
  }
}

extern "C" void kernel_launch(void* const* d_in, const int* in_sizes, int n_in,
                              void* d_out, int out_size, void* d_ws, size_t ws_size,
                              hipStream_t stream) {
  (void)in_sizes; (void)n_in; (void)out_size; (void)ws_size;
  const float* x     = (const float*)d_in[0];
  const float* w_qkv = (const float*)d_in[1];
  const float* b_qkv = (const float*)d_in[2];
  const float* w_out = (const float*)d_in[3];
  const float* b_out = (const float*)d_in[4];
  const float* ln_w  = (const float*)d_in[5];
  const float* ln_b  = (const float*)d_in[6];
  float*    out  = (float*)d_out;
  unsigned* wsws = (unsigned*)d_ws;      // 524288 B of packed bf16 weights

  // Repack weights into WMMA B-fragment order (runs every call; L2-resident).
  swin1d_pack_weights<<<(QKV_PAIRS + WOUT_PAIRS) / 256, 256, 0, stream>>>(
      w_qkv, w_out, wsws);

  (void)hipFuncSetAttribute((const void*)swin1d_fused_kernel,
                            hipFuncAttributeMaxDynamicSharedMemorySize,
                            SMEM_BYTES);
  const int blocks = 32 * (NSEQ / MTOK);   // 4096 tiles of 64 tokens
  swin1d_fused_kernel<<<blocks, 256, SMEM_BYTES, stream>>>(
      x, wsws, b_qkv, b_out, ln_w, ln_b, out);
}
